// MoELayer_1013612282515
// MI455X (gfx1250) — compile-verified, matrix operand order checked by reading, not run on hardware
//
#include <hip/hip_runtime.h>
#include <hip/hip_bf16.h>

// ---------------- problem constants (from reference) ----------------
#define H_DIM 512
#define E_NUM 32
#define F_DIM 2048
#define M_NUM 3
#define FC    128         // F chunk per WMMA pass
#define MT    32          // tokens per tile (2 stacked 16-row WMMA M-tiles)
#define ENTROPY_THRESHOLD 1.3862943611198906f

typedef _Float16 v16h __attribute__((ext_vector_type(16)));
typedef _Float16 v8h  __attribute__((ext_vector_type(8)));
typedef float    v8f  __attribute__((ext_vector_type(8)));

// ---------------- WMMA fragment gathers (wave32 layouts, CDNA5 ISA) ----
// For both A (16x32, row-major [M][K]) and B (32x16, stored [N][K]) each lane's
// 16 halves are two contiguous 8-half runs along K: [kb, kb+8) and [kb+16, kb+24),
// kb = k0 + 8*(lane>>4).  Two aligned 128-bit LDS loads per fragment.
static __device__ __forceinline__ v16h load_frag_rowK(const _Float16* __restrict__ base,
                                                      int rowstride, int row, int k0,
                                                      int lane) {
  int kb = k0 + ((lane >> 4) << 3);
  const _Float16* p = base + row * rowstride + kb;
  v8h lo = *(const v8h*)p;          // K = kb .. kb+7   (16B aligned)
  v8h hi = *(const v8h*)(p + 16);   // K = kb+16 .. kb+23
  return __builtin_shufflevector(lo, hi, 0, 1, 2, 3, 4, 5, 6, 7,
                                 8, 9, 10, 11, 12, 13, 14, 15);
}

static __device__ __forceinline__ v16h load_A_frag(const _Float16* __restrict__ tile,
                                                   int k0, int stride, int lane) {
  return load_frag_rowK(tile, stride, lane & 15, k0, lane);      // row = M
}

// B tile stored [N][K] with row stride strideK
static __device__ __forceinline__ v16h load_B_frag(const _Float16* __restrict__ tile,
                                                   int k0, int n0, int strideK, int lane) {
  return load_frag_rowK(tile, strideK, n0 + (lane & 15), k0, lane);  // row = N
}

static __device__ __forceinline__ v8f wmma_f16(v16h a, v16h b, v8f c) {
  return __builtin_amdgcn_wmma_f32_16x16x32_f16(false, a, false, b, (short)0, c,
                                                false, false);
}

// ---------------- elementwise f32 -> f16 convert -----------------------------
__global__ void convert_f32_f16(const float* __restrict__ src, _Float16* __restrict__ dst,
                                long long n) {
  long long i = (long long)blockIdx.x * blockDim.x + threadIdx.x;
  long long s = (long long)gridDim.x * blockDim.x;
  for (; i < n; i += s) dst[i] = (_Float16)src[i];
}

// ---------------- f32 -> f16 convert with transpose --------------------------
// src: [Z][R][C] f32, dst: [Z][C][R] f16.  32x32 LDS tile, block = 32x8.
__global__ __launch_bounds__(256) void transpose_convert_kernel(
    const float* __restrict__ src, _Float16* __restrict__ dst, int R, int C) {
  __shared__ float tile[32][33];
  const size_t mat = (size_t)R * C;
  const float* s = src + (size_t)blockIdx.z * mat;
  _Float16*    d = dst + (size_t)blockIdx.z * mat;
  const int c0 = blockIdx.x * 32, r0 = blockIdx.y * 32;
  const int tx = threadIdx.x & 31, ty = threadIdx.x >> 5;
#pragma unroll
  for (int i = 0; i < 4; ++i)
    tile[ty + i * 8][tx] = s[(size_t)(r0 + ty + i * 8) * C + c0 + tx];
  __syncthreads();
#pragma unroll
  for (int i = 0; i < 4; ++i)
    d[(size_t)(c0 + ty + i * 8) * R + r0 + tx] = (_Float16)tile[tx][ty + i * 8];
}

// ---------------- small utility kernels --------------------------------------
__global__ void zero_counts_kernel(int* counts) {
  if (threadIdx.x < E_NUM) counts[threadIdx.x] = 0;
}

__global__ void scan_kernel(const int* __restrict__ counts, int* __restrict__ offs,
                            int* __restrict__ cursor) {
  if (threadIdx.x == 0 && blockIdx.x == 0) {
    int a = 0;
    for (int e = 0; e < E_NUM; ++e) { offs[e] = a; cursor[e] = a; a += counts[e]; }
    offs[E_NUM] = a;
  }
}

__global__ void bucket_kernel(const int* __restrict__ expert_of, int* cursor,
                              int* __restrict__ bucket, int N) {
  int t = blockIdx.x * blockDim.x + threadIdx.x;
  if (t < N) {
    int e = expert_of[t];
    int pos = atomicAdd(&cursor[e], 1);
    bucket[pos] = t;   // order within expert irrelevant: each token's row is independent
  }
}

// ---------------- gating: one wave32 per token -------------------------------
__global__ __launch_bounds__(256) void gate_kernel(
    const float* __restrict__ x, const float* __restrict__ gate_w,
    const float* __restrict__ gate_b, const int* __restrict__ mmask,
    int* __restrict__ expert_of, int* __restrict__ counts,
    float* __restrict__ partials /* [nblocks][102] : 96 marg + 3 ent + 3 cnt */) {
  __shared__ float marg_w[8][M_NUM][E_NUM];
  __shared__ float ent_w[8][M_NUM];
  __shared__ float cnt_w[8][M_NUM];

  const int tid  = threadIdx.x;
  const int lane = tid & 31;
  const int wv   = tid >> 5;
  const int token = blockIdx.x * 8 + wv;

  const float* xr = x + (size_t)token * H_DIM;
  float acc = gate_b[lane];
  for (int h = 0; h < H_DIM; ++h)
    acc = fmaf(xr[h], gate_w[h * E_NUM + lane], acc);

  // softmax across the 32 lanes
  float mx = acc;
  for (int o = 16; o >= 1; o >>= 1) mx = fmaxf(mx, __shfl_xor(mx, o, 32));
  float p = expf(acc - mx);
  float s = p;
  for (int o = 16; o >= 1; o >>= 1) s += __shfl_xor(s, o, 32);
  p /= s;

  // token entropy
  float ent = -p * logf(p + 1e-10f);
  for (int o = 16; o >= 1; o >>= 1) ent += __shfl_xor(ent, o, 32);

  // argmax expert (lowest index tie-break, matches top_k)
  float bv = acc; int bi = lane;
  for (int o = 16; o >= 1; o >>= 1) {
    float ov = __shfl_xor(bv, o, 32);
    int   oi = __shfl_xor(bi, o, 32);
    if (ov > bv || (ov == bv && oi < bi)) { bv = ov; bi = oi; }
  }

#pragma unroll
  for (int m = 0; m < M_NUM; ++m) {
    int msk = mmask[token * M_NUM + m];
    marg_w[wv][m][lane] = msk ? p : 0.0f;
    if (lane == 0) {
      ent_w[wv][m] = msk ? ent : 0.0f;
      cnt_w[wv][m] = (float)msk;
    }
  }
  if (lane == 0) {
    expert_of[token] = bi;
    atomicAdd(&counts[bi], 1);
  }
  __syncthreads();

  float* pb = partials + (size_t)blockIdx.x * 102;
  if (tid < 96) {
    int m = tid >> 5, e = tid & 31;
    float sum = 0.f;
    for (int w = 0; w < 8; ++w) sum += marg_w[w][m][e];
    pb[tid] = sum;
  } else if (tid < 99) {
    int m = tid - 96; float sum = 0.f;
    for (int w = 0; w < 8; ++w) sum += ent_w[w][m];
    pb[tid] = sum;
  } else if (tid < 102) {
    int m = tid - 99; float sum = 0.f;
    for (int w = 0; w < 8; ++w) sum += cnt_w[w][m];
    pb[tid] = sum;
  }
}

__global__ void reduce_partials_kernel(const float* __restrict__ partials,
                                       float* __restrict__ reduced, int nblocks) {
  int i = threadIdx.x;
  if (i < 102) {
    float s = 0.f;
    for (int b = 0; b < nblocks; ++b) s += partials[(size_t)b * 102 + i];
    reduced[i] = s;
  }
}

__global__ void aux_kernel(const float* __restrict__ reduced, float* __restrict__ out_aux) {
  if (threadIdx.x == 0 && blockIdx.x == 0) {
    float total = 0.f, nvalid = 0.f; bool any = false;
    for (int m = 0; m < M_NUM; ++m) {
      float cnt  = reduced[99 + m];
      float safe = fmaxf(cnt, 1.0f);
      float local = reduced[96 + m] / safe;
      float ge = 0.f;
      for (int e = 0; e < E_NUM; ++e) {
        float q = reduced[m * E_NUM + e] / safe;
        ge -= q * logf(q + 1e-10f);
      }
      float gl = fmaxf(0.f, ENTROPY_THRESHOLD + ge);
      if (cnt > 0.f) { total += local + gl; nvalid += 1.f; any = true; }
    }
    *out_aux = any ? total / fmaxf(nvalid, 1.f) : 0.f;
  }
}

// ---------------- WMMA expert FFN --------------------------------------------
// Grid (E_NUM, 8). Block = 256 threads = 8 waves. 32-token M-tiles (two stacked
// 16x16 WMMA tiles) double A/B reuse: each B fragment feeds 2 WMMAs, each staged
// weight tile serves 32 tokens.  Weight tiles in LDS in [N][K] order (from
// pre-transposed w1t/w2t) so all fragments are 2 aligned ds_load_b128s.
// LDS: Xs 32K + W1s 128K + W2s 128K + Hs 8K ~= 296KB (<320KB WGP budget).
#define SMEM_HALVES (MT * H_DIM + FC * H_DIM + H_DIM * FC + MT * FC)
#define SMEM_BYTES  (SMEM_HALVES * 2 + MT * 4)

__global__ __launch_bounds__(256) void ffn_wmma_kernel(
    const _Float16* __restrict__ xh,
    const _Float16* __restrict__ w1t,   // [E][F][H]  (w1 transposed)
    const _Float16* __restrict__ w2t,   // [E][H][F]  (w2 transposed)
    const float* __restrict__ b1, const float* __restrict__ b2,
    const int* __restrict__ bucket, const int* __restrict__ offs,
    const int* __restrict__ counts, float* __restrict__ out) {
  extern __shared__ char smem[];
  _Float16* Xs  = (_Float16*)smem;                                  // [32 M][512 K]
  _Float16* W1s = Xs + MT * H_DIM;                                  // [128 N][512 K]
  _Float16* W2s = W1s + FC * H_DIM;                                 // [512 N][128 K]
  _Float16* Hs  = W2s + H_DIM * FC;                                 // [32 M][128 K]
  int*      toks = (int*)(Hs + MT * FC);                            // [32]

  const int e    = blockIdx.x;
  const int tid  = threadIdx.x;
  const int lane = tid & 31;
  const int wv   = tid >> 5;                 // 0..7
  const int cnt  = counts[e];
  const int base = offs[e];
  const int ntiles = (cnt + MT - 1) / MT;
  const _Float16* w1e = w1t + (size_t)e * H_DIM * F_DIM;
  const _Float16* w2e = w2t + (size_t)e * F_DIM * H_DIM;
  const int hi = lane >> 4;
  const int nl = lane & 15;
  const v8f vzero = {0.f, 0.f, 0.f, 0.f, 0.f, 0.f, 0.f, 0.f};

  for (int t = blockIdx.y; t < ntiles; t += gridDim.y) {
    __syncthreads();  // toks/Xs free from previous tile
    if (tid < MT) {
      int idx = t * MT + tid;
      toks[tid] = (idx < cnt) ? bucket[base + idx] : -1;
    }
    __syncthreads();

    // gather 32 token rows of x (f16) into LDS
    {
      int row = tid >> 3;              // 32 rows, 8 threads per row
      int c0  = (tid & 7) * 64;        // 64 halves each
      int g   = toks[row];
      _Float16* dst = Xs + row * H_DIM + c0;
      if (g >= 0) {
        const _Float16* src = xh + (size_t)g * H_DIM + c0;
#pragma unroll
        for (int i = 0; i < 8; ++i) ((float4*)dst)[i] = ((const float4*)src)[i];
      } else {
#pragma unroll
        for (int i = 0; i < 64; ++i) dst[i] = (_Float16)0.f;
      }
    }

    v8f yacc[2][4];   // [m-tile][n-tile]; wave owns Y cols [wv*64, wv*64+64)
#pragma unroll
    for (int p = 0; p < 2; ++p)
#pragma unroll
      for (int j = 0; j < 4; ++j) yacc[p][j] = vzero;

    for (int fc = 0; fc < F_DIM / FC; ++fc) {
      __syncthreads();  // previous stage2 done; Xs gathered (first iter)

      // stage W1 tile: w1t rows [fc*128, +128), each 512 contiguous halves -> W1s [128][512]
      {
        int n  = tid >> 1;               // 0..127
        int c0 = (tid & 1) * 256;        // 256 halves each
        const _Float16* src = w1e + (size_t)(fc * FC + n) * H_DIM + c0;
        _Float16* dst = W1s + n * H_DIM + c0;
#pragma unroll
        for (int i = 0; i < 32; ++i) ((float4*)dst)[i] = ((const float4*)src)[i];
      }
      // stage W2 tile: w2t rows 0..511, cols [fc*128, +128) contiguous -> W2s [512][128]
#pragma unroll
      for (int r = 0; r < 2; ++r) {
        int n = tid * 2 + r;             // 0..511
        const _Float16* src = w2e + (size_t)n * F_DIM + fc * FC;
        _Float16* dst = W2s + n * FC;
#pragma unroll
        for (int i = 0; i < 16; ++i) ((float4*)dst)[i] = ((const float4*)src)[i];
      }
      if (fc + 1 < F_DIM / FC) {  // gfx1250 global_prefetch_b8 for next chunk
        __builtin_prefetch(w1e + (size_t)((fc + 1) * FC + (tid >> 1)) * H_DIM + (tid & 1) * 256, 0, 1);
        __builtin_prefetch(w2e + (size_t)(tid * 2) * F_DIM + (fc + 1) * FC, 0, 1);
      }
      __syncthreads();

      // stage 1: wave wv computes H-chunk cols [wv*16, wv*16+16) for 32 rows, K = H_DIM
      // (B fragment shared by the two M-tiles; loads pipelined one k-step ahead)
      {
        v8f hacc0 = vzero, hacc1 = vzero;
        v16h a0 = load_A_frag(Xs, 0, H_DIM, lane);
        v16h a1 = load_A_frag(Xs + 16 * H_DIM, 0, H_DIM, lane);
        v16h bb = load_B_frag(W1s, 0, wv * 16, H_DIM, lane);
#pragma unroll
        for (int ks = 0; ks < H_DIM / 32; ++ks) {
          v16h a0n = a0, a1n = a1, bn = bb;
          if (ks + 1 < H_DIM / 32) {
            a0n = load_A_frag(Xs, (ks + 1) * 32, H_DIM, lane);
            a1n = load_A_frag(Xs + 16 * H_DIM, (ks + 1) * 32, H_DIM, lane);
            bn  = load_B_frag(W1s, (ks + 1) * 32, wv * 16, H_DIM, lane);
          }
          hacc0 = wmma_f16(a0, bb, hacc0);
          hacc1 = wmma_f16(a1, bb, hacc1);
          a0 = a0n; a1 = a1n; bb = bn;
        }
        float bvv = b1[e * F_DIM + fc * FC + wv * 16 + nl];
        int col = wv * 16 + nl;
#pragma unroll
        for (int v = 0; v < 8; ++v) {
          int m = v + 8 * hi;
          float v0 = hacc0[v] + bvv;
          v0 = 0.5f * v0 * (1.0f + erff(v0 * 0.70710678118654752f));  // exact gelu
          Hs[m * FC + col] = (_Float16)v0;
          float v1 = hacc1[v] + bvv;
          v1 = 0.5f * v1 * (1.0f + erff(v1 * 0.70710678118654752f));
          Hs[(16 + m) * FC + col] = (_Float16)v1;
        }
      }
      __syncthreads();

      // stage 2: Y[32x512] += Hs[32x128] @ W2[128x512]; wave wv owns cols wv*64..+64
      // (A fragments shared across 4 n-tiles; each B fragment feeds 2 WMMAs)
      {
        v16h a0 = load_A_frag(Hs, 0, FC, lane);
        v16h a1 = load_A_frag(Hs + 16 * FC, 0, FC, lane);
        v16h bb = load_B_frag(W2s, 0, wv * 64, FC, lane);
#pragma unroll
        for (int ks = 0; ks < FC / 32; ++ks) {
          v16h a0n = a0, a1n = a1;
          if (ks + 1 < FC / 32) {
            a0n = load_A_frag(Hs, (ks + 1) * 32, FC, lane);
            a1n = load_A_frag(Hs + 16 * FC, (ks + 1) * 32, FC, lane);
          }
#pragma unroll
          for (int j = 0; j < 4; ++j) {
            v16h bn = bb;
            if (j + 1 < 4)
              bn = load_B_frag(W2s, ks * 32, wv * 64 + (j + 1) * 16, FC, lane);
            else if (ks + 1 < FC / 32)
              bn = load_B_frag(W2s, (ks + 1) * 32, wv * 64, FC, lane);
            yacc[0][j] = wmma_f16(a0, bb, yacc[0][j]);
            yacc[1][j] = wmma_f16(a1, bb, yacc[1][j]);
            bb = bn;
          }
          a0 = a0n; a1 = a1n;
        }
      }
    }

    // scatter Y (+b2) to original token rows
#pragma unroll
    for (int j = 0; j < 4; ++j) {
      int col = wv * 64 + j * 16 + nl;
      float bvv = b2[e * H_DIM + col];
#pragma unroll
      for (int p = 0; p < 2; ++p) {
#pragma unroll
        for (int v = 0; v < 8; ++v) {
          int m = p * 16 + v + 8 * hi;
          int g = toks[m];
          if (g >= 0) out[(size_t)g * H_DIM + col] = yacc[p][j][v] + bvv;
        }
      }
    }
  }
}

// ---------------- host launcher ----------------------------------------------
extern "C" void kernel_launch(void* const* d_in, const int* in_sizes, int n_in,
                              void* d_out, int out_size, void* d_ws, size_t ws_size,
                              hipStream_t stream) {
  (void)n_in; (void)out_size; (void)ws_size;
  const float* x      = (const float*)d_in[0];
  const float* gate_w = (const float*)d_in[1];
  const float* gate_b = (const float*)d_in[2];
  const float* w1     = (const float*)d_in[3];
  const float* b1     = (const float*)d_in[4];
  const float* w2     = (const float*)d_in[5];
  const float* b2     = (const float*)d_in[6];
  const int*   mmask  = (const int*)d_in[7];
  float* out = (float*)d_out;

  const int N = in_sizes[0] / H_DIM;  // 4096 tokens

  // ---- workspace layout ----
  char* ws = (char*)d_ws;
  const size_t szW = (size_t)E_NUM * H_DIM * F_DIM * sizeof(_Float16);  // 64 MB each
  _Float16* w1t = (_Float16*)ws;                 ws += szW;
  _Float16* w2t = (_Float16*)ws;                 ws += szW;
  _Float16* xh  = (_Float16*)ws;                 ws += (size_t)N * H_DIM * sizeof(_Float16);
  int* expert_of = (int*)ws;                     ws += (size_t)N * sizeof(int);
  int* bucket    = (int*)ws;                     ws += (size_t)N * sizeof(int);
  int* counts    = (int*)ws;                     ws += 256;
  int* offs      = (int*)ws;                     ws += 256;
  int* cursor    = (int*)ws;                     ws += 256;
  float* partials = (float*)ws;                  ws += (size_t)(N / 8) * 102 * sizeof(float);
  float* reduced  = (float*)ws;

  // transpose-convert weights: w1 [E][H][F] -> w1t [E][F][H]; w2 [E][F][H] -> w2t [E][H][F]
  {
    dim3 g1(F_DIM / 32, H_DIM / 32, E_NUM);
    transpose_convert_kernel<<<g1, 256, 0, stream>>>(w1, w1t, H_DIM, F_DIM);
    dim3 g2(H_DIM / 32, F_DIM / 32, E_NUM);
    transpose_convert_kernel<<<g2, 256, 0, stream>>>(w2, w2t, F_DIM, H_DIM);
  }
  convert_f32_f16<<<512, 256, 0, stream>>>(x, xh, (long long)N * H_DIM);

  zero_counts_kernel<<<1, 32, 0, stream>>>(counts);
  gate_kernel<<<N / 8, 256, 0, stream>>>(x, gate_w, gate_b, mmask,
                                         expert_of, counts, partials);
  reduce_partials_kernel<<<1, 128, 0, stream>>>(partials, reduced, N / 8);
  scan_kernel<<<1, 1, 0, stream>>>(counts, offs, cursor);
  bucket_kernel<<<(N + 255) / 256, 256, 0, stream>>>(expert_of, cursor, bucket, N);

  dim3 grid(E_NUM, 8);
  ffn_wmma_kernel<<<grid, 256, SMEM_BYTES, stream>>>(xh, w1t, w2t, b1, b2,
                                                     bucket, offs, counts, out);

  aux_kernel<<<1, 1, 0, stream>>>(reduced, out + (size_t)N * H_DIM);
}
